// attention_39685497815301
// MI455X (gfx1250) — matrix operand order repaired and checked by reading, not compiled, not run on hardware
//
#include <hip/hip_runtime.h>

// ---------------------------------------------------------------------------
// MI455X (gfx1250, wave32) implementation.
//   K1 k_proj : f/g/h = relu(x@W+b) via v_wmma_f32_16x16x32_bf16; weights
//               pulled into LDS by the Tensor Data Mover (tensor_load_to_lds),
//               outputs stored pre-packed in their consumer's operand layout.
//   K2 k_attn : flash attention (online softmax) + wr projection + channel
//               mean, fused per 16-query tile per wave; wr preloaded by TDM
//               overlapped with the whole flash loop. 8 WMMA / 32 keys.
//   K3 k_ima1 : flat@w1 — HBM bound (w1 = 67 MB), float4 streaming + prefetch.
//   K4 k_final: tiny MLPs, argmax one-hot with batch flip.
// ---------------------------------------------------------------------------

typedef __attribute__((ext_vector_type(16))) __bf16 v16bf;
typedef __attribute__((ext_vector_type(8)))  float  v8f;
typedef unsigned __attribute__((ext_vector_type(4))) uint4v;
typedef int      __attribute__((ext_vector_type(4))) int4v;
typedef int      __attribute__((ext_vector_type(8))) int8v;

union BF16x16 { unsigned u[8]; v16bf v; };

// bf16 via truncation: cheap (2 VALU) and plenty for mean->argmax consumers.
__device__ __forceinline__ unsigned short bf16u(float x) {
    return (unsigned short)(__builtin_bit_cast(unsigned, x) >> 16);
}
__device__ __forceinline__ unsigned pk(float lo, float hi) {
    return (__builtin_bit_cast(unsigned, hi) & 0xFFFF0000u) |
           (__builtin_bit_cast(unsigned, lo) >> 16);
}

// ds_swizzle_b32 xor-butterfly (group-of-32 mode: offset[14:10]=xor, [4:0]=and)
template <int MASK>
__device__ __forceinline__ float sxor(float v) {
    int r = __builtin_amdgcn_ds_swizzle(__builtin_bit_cast(int, v),
                                        (MASK << 10) | 0x1F);
    return __builtin_bit_cast(float, r);
}
__device__ __forceinline__ float redmax16(float t) {
    t = fmaxf(t, sxor<1>(t)); t = fmaxf(t, sxor<2>(t));
    t = fmaxf(t, sxor<4>(t)); t = fmaxf(t, sxor<8>(t));
    return t;
}
__device__ __forceinline__ float redsum16(float t) {
    t += sxor<1>(t); t += sxor<2>(t); t += sxor<4>(t); t += sxor<8>(t);
    return t;
}

#define WMMA_BF16(A, B, C) \
    __builtin_amdgcn_wmma_f32_16x16x32_bf16(false, (A), false, (B), (short)0, (C), false, false)

// ---------------------------------------------------------------------------
// TDM: 1-D fp32 tile (nElem floats) global -> LDS. D# per ISA 08 §8:
//  g0: count=1 | lds_addr | global_addr(57b) | type=2
//  g1: data_size=4B; tensor_dim0=tile_dim0=nElem; tensor_dim1=tile_dim1=1;
//      tensor_dim0_stride=nElem. All values wave-uniform -> SGPR descriptor.
// ---------------------------------------------------------------------------
__device__ __forceinline__ void tdm_load_1d(unsigned ldsByteAddr,
                                            const float* gaddr,
                                            unsigned nElem) {
    unsigned long long ga = (unsigned long long)(size_t)gaddr;
    uint4v g0;
    g0.x = 1u;                                            // count=1 (user D#)
    g0.y = ldsByteAddr;                                   // lds_addr
    g0.z = (unsigned)(ga & 0xFFFFFFFFu);                  // global_addr[31:0]
    g0.w = (unsigned)((ga >> 32) & 0x01FFFFFFu) | (2u << 30); // [56:32] | type=2
    int8v g1;
    g1[0] = 0x20000;                                      // data_size=4B, mask=0
    g1[1] = (int)((nElem & 0xFFFFu) << 16);               // tensor_dim0 lo16
    g1[2] = (int)(((nElem >> 16) & 0xFFFFu) | (1u << 16));// dim0 hi16 | dim1=1
    g1[3] = (int)((nElem & 0xFFFFu) << 16);               // tile_dim0
    g1[4] = 1;                                            // tile_dim1=1
    g1[5] = (int)nElem;                                   // dim0_stride lo32
    g1[6] = 0;
    g1[7] = 0;
    int4v gz = {0, 0, 0, 0};
#if defined(__clang_major__) && (__clang_major__ >= 23)
    int8v gz8 = {0, 0, 0, 0, 0, 0, 0, 0};
    __builtin_amdgcn_tensor_load_to_lds(g0, g1, gz, gz, gz8, 0);
#else
    __builtin_amdgcn_tensor_load_to_lds(g0, g1, gz, gz, 0);
#endif
}

// Problem constants
#define NB   4
#define NPIX 4096            // H*W
#define NC   64              // channels
#define NROW (NB * NPIX)     // 16384

// ---------------------------------------------------------------------------
// K1: projections. 1 wave = 16 pixel rows x 64 channels x {wf,wg,wh}.
// Operand k order: contiguous k = 16*half + s in both A and B (see comments in round-0 version).
// ---------------------------------------------------------------------------
__global__ __launch_bounds__(128) void k_proj(
    const float* __restrict__ x,
    const float* __restrict__ wf, const float* __restrict__ bfv,
    const float* __restrict__ wg, const float* __restrict__ bgv,
    const float* __restrict__ wh, const float* __restrict__ bhv,
    unsigned short* __restrict__ gOut,   // [NROW][64] bf16 row-major
    unsigned*       __restrict__ fpair,  // [B][32][4096] dwords (f[n][2c],f[n][2c+1])
    unsigned*       __restrict__ hpack,  // [B][2048][64] dwords (h[2j][c],h[2j+1][c])
    float*          __restrict__ xmean)  // [NROW]
{
    __shared__ float sW[3 * 4096];                       // 48 KB: wf|wg|wh
    const int lane = threadIdx.x & 31;
    const int wv   = threadIdx.x >> 5;
    const int tile = blockIdx.x * 4 + wv;
    const int q0   = tile * 16;
    const int half = lane >> 4;
    const int col  = lane & 15;
    const int row  = q0 + col;
    const int bIdx = q0 >> 12;
    const int jloc = q0 & (NPIX - 1);

    // Wave 0 kicks the TDM for all three weight matrices (async, TENSORcnt).
    const unsigned ldsW = (unsigned)(size_t)(void*)sW;   // low 32b = LDS offset
    if (threadIdx.x < 32) {
        tdm_load_1d(ldsW,         wf, 4096);
        tdm_load_1d(ldsW + 16384, wg, 4096);
        tdm_load_1d(ldsW + 32768, wh, 4096);
    }

    // Overlap: load x rows as two bf16 A operands + channel mean.
    BF16x16 A0, A1;
    float rsum = 0.f;
#pragma unroll
    for (int v = 0; v < 8; ++v) {
        int k = 16 * half + 2 * v;
        float a0 = x[row * NC + k],      a1 = x[row * NC + k + 1];
        float a2 = x[row * NC + 32 + k], a3 = x[row * NC + 33 + k];
        A0.u[v] = pk(a0, a1);
        A1.u[v] = pk(a2, a3);
        rsum += a0 + a1 + a2 + a3;
    }
    rsum += sxor<16>(rsum);                              // both halves -> 64 ch
    if (lane < 16) xmean[row] = rsum * (1.f / 64.f);

    if (threadIdx.x < 32) __builtin_amdgcn_s_wait_tensorcnt(0);
    __syncthreads();                                     // weights visible in LDS

    const float* Bs[3] = { bfv, bgv, bhv };
    unsigned short* fp16 = (unsigned short*)fpair;
#pragma unroll
    for (int s = 0; s < 3; ++s) {
        const float* W  = sW + s * 4096;
        const float* Bv = Bs[s];
#pragma unroll
        for (int nt = 0; nt < 4; ++nt) {
            BF16x16 B0, B1;
#pragma unroll
            for (int v = 0; v < 8; ++v) {
                int k  = 16 * half + 2 * v;
                int cc = nt * 16 + col;
                B0.u[v] = pk(W[k * NC + cc],        W[(k + 1) * NC + cc]);
                B1.u[v] = pk(W[(k + 32) * NC + cc], W[(k + 33) * NC + cc]);
            }
            v8f c = {};
            c = WMMA_BF16(A0.v, B0.v, c);
            c = WMMA_BF16(A1.v, B1.v, c);
            float bias = Bv[nt * 16 + col];
            float y[8];
#pragma unroll
            for (int r = 0; r < 8; ++r) {
                float t = c[r] + bias;
                y[r] = t > 0.f ? t : 0.f;
            }
            int ch = nt * 16 + col;
            if (s == 1) {                 // g row-major bf16: plain b16 stores
#pragma unroll
                for (int r = 0; r < 8; ++r) {
                    int pix = q0 + r + (half ? 8 : 0);
                    gOut[pix * NC + ch] = bf16u(y[r]);
                }
            } else if (s == 0) {          // fpair viewed as u16: [b][ch/2][pix][ch&1]
#pragma unroll
                for (int r = 0; r < 8; ++r) {
                    int pixl = jloc + r + (half ? 8 : 0);
                    fp16[(bIdx * 131072 + (ch >> 1) * NPIX + pixl) * 2 + (ch & 1)] =
                        bf16u(y[r]);
                }
            } else {                      // hpack: row pairs live in one lane
#pragma unroll
                for (int t = 0; t < 4; ++t) {
                    int pixl = jloc + 2 * t + (half ? 8 : 0);
                    hpack[bIdx * 131072 + (pixl >> 1) * NC + ch] = pk(y[2 * t], y[2 * t + 1]);
                }
            }
        }
    }
}

// ---------------------------------------------------------------------------
// K2: fused flash attention + wr projection + channel-mean blend.
// wr is TDM-preloaded into LDS, wait deferred until after the flash loop.
// ---------------------------------------------------------------------------
__global__ __launch_bounds__(256) void k_attn(
    const unsigned short* __restrict__ gIn,
    const unsigned*       __restrict__ fpair,
    const unsigned*       __restrict__ hpack,
    const float* __restrict__ wr, const float* __restrict__ br,
    const float* __restrict__ xmean,
    float* __restrict__ mOut)
{
    __shared__ unsigned short lds[8 * 16 * 64];          // 16 KB: 2 KB / wave
    __shared__ float swr[4096];                          // 16 KB: wr via TDM
    const int lane = threadIdx.x & 31;
    const int wv   = threadIdx.x >> 5;
    const int qt   = blockIdx.x * 8 + wv;
    const int q0   = qt * 16;
    const int b    = q0 >> 12;
    const int half = lane >> 4;
    const int col  = lane & 15;
    unsigned short* wlds = lds + wv * 1024;

    if (threadIdx.x < 32)
        tdm_load_1d((unsigned)(size_t)(void*)swr, wr, 4096);  // overlaps flash loop

    // Resident A operand: 16 query rows of g.
    BF16x16 Ag0, Ag1;
    {
        int row = q0 + col;
#pragma unroll
        for (int v = 0; v < 8; ++v) {
            int k = 16 * half + 2 * v;
            Ag0.u[v] = *reinterpret_cast<const unsigned*>(gIn + row * NC + k);
            Ag1.u[v] = *reinterpret_cast<const unsigned*>(gIn + row * NC + 32 + k);
        }
    }

    v8f O0 = {}, O1 = {}, O2 = {}, O3 = {};
    float M[8], L[8];
#pragma unroll
    for (int r = 0; r < 8; ++r) { M[r] = -1e30f; L[r] = 0.f; }

    const unsigned* fpB = fpair + b * 131072;
    const unsigned* hpB = hpack + b * 131072;

    for (int j0 = 0; j0 < NPIX; j0 += 32) {
        BF16x16 Bf;
        v8f S0 = {}, S1 = {};
#pragma unroll
        for (int v = 0; v < 8; ++v) Bf.u[v] = fpB[(8 * half + v) * NPIX + j0 + col];
        S0 = WMMA_BF16(Ag0.v, Bf.v, S0);
#pragma unroll
        for (int v = 0; v < 8; ++v) Bf.u[v] = fpB[(16 + 8 * half + v) * NPIX + j0 + col];
        S0 = WMMA_BF16(Ag1.v, Bf.v, S0);
#pragma unroll
        for (int v = 0; v < 8; ++v) Bf.u[v] = fpB[(8 * half + v) * NPIX + j0 + 16 + col];
        S1 = WMMA_BF16(Ag0.v, Bf.v, S1);
#pragma unroll
        for (int v = 0; v < 8; ++v) Bf.u[v] = fpB[(16 + 8 * half + v) * NPIX + j0 + 16 + col];
        S1 = WMMA_BF16(Ag1.v, Bf.v, S1);

        // online softmax: rows of the D tile live across 16 lanes
#pragma unroll
        for (int r = 0; r < 8; ++r) {
            float t  = redmax16(fmaxf(S0[r], S1[r]));
            float Mn = fmaxf(M[r], t);
            float sc = __expf(M[r] - Mn);
            float p0 = __expf(S0[r] - Mn);
            float p1 = __expf(S1[r] - Mn);
            float ps = redsum16(p0 + p1);
            L[r] = L[r] * sc + ps;
            M[r] = Mn;
            O0[r] *= sc; O1[r] *= sc; O2[r] *= sc; O3[r] *= sc;
            int prow = r + (half ? 8 : 0);
            wlds[prow * 32 + col]      = bf16u(p0);      // stage P [16][32]
            wlds[prow * 32 + 16 + col] = bf16u(p1);
        }

        // D-layout -> A-layout transpose through private LDS (same-wave DS ops
        // are in-order, no barrier needed)
        BF16x16 Pa;
#pragma unroll
        for (int v = 0; v < 8; ++v) {
            int k = 16 * half + 2 * v;
            Pa.u[v] = *reinterpret_cast<const unsigned*>(wlds + col * 32 + k);
        }

        BF16x16 Bh;
        const int jp = (j0 >> 1) + 8 * half;
#pragma unroll
        for (int v = 0; v < 8; ++v) Bh.u[v] = hpB[(jp + v) * NC + col];
        O0 = WMMA_BF16(Pa.v, Bh.v, O0);
#pragma unroll
        for (int v = 0; v < 8; ++v) Bh.u[v] = hpB[(jp + v) * NC + 16 + col];
        O1 = WMMA_BF16(Pa.v, Bh.v, O1);
#pragma unroll
        for (int v = 0; v < 8; ++v) Bh.u[v] = hpB[(jp + v) * NC + 32 + col];
        O2 = WMMA_BF16(Pa.v, Bh.v, O2);
#pragma unroll
        for (int v = 0; v < 8; ++v) Bh.u[v] = hpB[(jp + v) * NC + 48 + col];
        O3 = WMMA_BF16(Pa.v, Bh.v, O3);
    }

    // wr must now be in LDS for every wave
    if (threadIdx.x < 32) __builtin_amdgcn_s_wait_tensorcnt(0);
    __syncthreads();

    // epilogue: O/L, o = relu(O@wr+br), channel mean, blend with mean(x)
#pragma unroll
    for (int r = 0; r < 8; ++r) {
        float inv = 1.f / L[r];
        int prow = r + (half ? 8 : 0);
        wlds[prow * 64 + col]      = bf16u(O0[r] * inv);
        wlds[prow * 64 + 16 + col] = bf16u(O1[r] * inv);
        wlds[prow * 64 + 32 + col] = bf16u(O2[r] * inv);
        wlds[prow * 64 + 48 + col] = bf16u(O3[r] * inv);
    }
    BF16x16 Ao0, Ao1;
#pragma unroll
    for (int v = 0; v < 8; ++v) {
        int k = 16 * half + 2 * v;
        Ao0.u[v] = *reinterpret_cast<const unsigned*>(wlds + col * 64 + k);
        Ao1.u[v] = *reinterpret_cast<const unsigned*>(wlds + col * 64 + 32 + k);
    }
    float csum[8] = {0.f, 0.f, 0.f, 0.f, 0.f, 0.f, 0.f, 0.f};
#pragma unroll
    for (int ct = 0; ct < 4; ++ct) {
        BF16x16 B0, B1;
#pragma unroll
        for (int v = 0; v < 8; ++v) {
            int k  = 16 * half + 2 * v;
            int cc = ct * 16 + col;
            B0.u[v] = pk(swr[k * NC + cc],        swr[(k + 1) * NC + cc]);
            B1.u[v] = pk(swr[(k + 32) * NC + cc], swr[(k + 33) * NC + cc]);
        }
        v8f c = {};
        c = WMMA_BF16(Ao0.v, B0.v, c);
        c = WMMA_BF16(Ao1.v, B1.v, c);
        float bias = br[ct * 16 + col];
#pragma unroll
        for (int r = 0; r < 8; ++r) csum[r] += fmaxf(c[r] + bias, 0.f);
    }
#pragma unroll
    for (int r = 0; r < 8; ++r) {
        float s = redsum16(csum[r]);
        if (col == 0) {
            int pix = q0 + r + (half ? 8 : 0);
            mOut[pix] = 0.1f * (s * (1.f / 64.f)) + 0.9f * xmean[pix];
        }
    }
}

// ---------------------------------------------------------------------------
// K3: ima1 = flat @ w1 (M=4, K=262144, N=64). HBM streaming of w1 with B128
// loads + global_prefetch_b8. ~67 MB -> ~2.9 us at 23.3 TB/s.
// ---------------------------------------------------------------------------
__global__ __launch_bounds__(256) void k_ima1(
    const float* __restrict__ x, const float* __restrict__ w1,
    float* __restrict__ ima1)                       // [4*64], pre-zeroed
{
    __shared__ float red[16][4][64];                // [kgroup][batch][channel]
    const int t  = threadIdx.x;
    const int c4 = (t & 15) * 4;
    const int kg = t >> 4;                          // 0..15
    const int KC = 4096;
    const int k0 = blockIdx.x * KC;
    float acc[4][4] = {};
    for (int k = k0 + kg; k < k0 + KC; k += 16) {
        if (((k - k0 - kg) & 1023) == 0 && k + 2048 < 262144)
            __builtin_prefetch(&w1[(k + 2048) * 64 + c4], 0, 1);
        float4 w = *reinterpret_cast<const float4*>(w1 + k * 64 + c4);
#pragma unroll
        for (int bb = 0; bb < 4; ++bb) {
            float xv = x[bb * 262144 + k];
            acc[bb][0] = fmaf(xv, w.x, acc[bb][0]);
            acc[bb][1] = fmaf(xv, w.y, acc[bb][1]);
            acc[bb][2] = fmaf(xv, w.z, acc[bb][2]);
            acc[bb][3] = fmaf(xv, w.w, acc[bb][3]);
        }
    }
#pragma unroll
    for (int bb = 0; bb < 4; ++bb)
#pragma unroll
        for (int j = 0; j < 4; ++j) red[kg][bb][c4 + j] = acc[bb][j];
    __syncthreads();
    const int bb = t >> 6, c = t & 63;
    float s = 0.f;
#pragma unroll
    for (int g2 = 0; g2 < 16; ++g2) s += red[g2][bb][c];
    atomicAdd(&ima1[bb * 64 + c], s);
}

__global__ void k_zero(float* __restrict__ p, int n) {
    int i = blockIdx.x * blockDim.x + threadIdx.x;
    if (i < n) p[i] = 0.f;
}

// ---------------------------------------------------------------------------
// K4: finalize (unchanged).
// ---------------------------------------------------------------------------
__global__ __launch_bounds__(512) void k_final(
    const float* __restrict__ ima1, const float* __restrict__ b1,
    const float* __restrict__ w2,   const float* __restrict__ b2,
    const float* __restrict__ m,
    const float* __restrict__ wc1,  const float* __restrict__ bc1,
    const float* __restrict__ wc2,  const float* __restrict__ bc2,
    float* __restrict__ out)                        // [592]
{
    __shared__ float sIma[4][64];
    __shared__ float sCd1[4][128];
    __shared__ float sCd2[4][20];
    __shared__ int   sAmax[4];
    const int t = threadIdx.x;
    if (t < 256) sIma[t >> 6][t & 63] = ima1[t] + b1[t & 63];
    __syncthreads();

    const int bb = t >> 7, d = t & 127;
    {
        float acc = b2[d];
        for (int k = 0; k < 64; ++k) acc = fmaf(sIma[bb][k], w2[k * 128 + d], acc);
        out[bb * 128 + d] = acc;
    }
    {
        float acc = bc1[d];
        for (int k = 0; k < 4096; ++k) acc = fmaf(m[bb * 4096 + k], wc1[k * 128 + d], acc);
        sCd1[bb][d] = fmaxf(acc, 0.f);
    }
    __syncthreads();
    if (t < 80) {
        int i = t / 20, j = t % 20;
        float acc = bc2[j];
        for (int k = 0; k < 128; ++k) acc = fmaf(sCd1[i][k], wc2[k * 20 + j], acc);
        sCd2[i][j] = acc;
    }
    __syncthreads();
    if (t < 4) {
        int best = 0; float bv = sCd2[t][0];
        for (int j = 1; j < 20; ++j) if (sCd2[t][j] > bv) { bv = sCd2[t][j]; best = j; }
        sAmax[t] = best;
    }
    __syncthreads();
    if (t < 80) {
        int i = t / 20, j = t % 20;
        out[512 + t] = (j == sAmax[3 - i]) ? 1.f : 0.f;   // flip(batch) + one_hot
    }
}

// ---------------------------------------------------------------------------
extern "C" void kernel_launch(void* const* d_in, const int* in_sizes, int n_in,
                              void* d_out, int out_size, void* d_ws, size_t ws_size,
                              hipStream_t stream) {
    const float* x   = (const float*)d_in[0];
    const float* wf  = (const float*)d_in[1];
    const float* bfv = (const float*)d_in[2];
    const float* wg  = (const float*)d_in[3];
    const float* bgv = (const float*)d_in[4];
    const float* wh  = (const float*)d_in[5];
    const float* bhv = (const float*)d_in[6];
    const float* wr  = (const float*)d_in[7];
    const float* br  = (const float*)d_in[8];
    const float* w1  = (const float*)d_in[9];
    const float* b1  = (const float*)d_in[10];
    const float* w2  = (const float*)d_in[11];
    const float* b2  = (const float*)d_in[12];
    const float* wc1 = (const float*)d_in[13];
    const float* bc1 = (const float*)d_in[14];
    const float* wc2 = (const float*)d_in[15];
    const float* bc2 = (const float*)d_in[16];
    float* out = (float*)d_out;

    char* ws = (char*)d_ws;
    unsigned short* gBuf  = (unsigned short*)(ws);                   // 2 MB
    unsigned*       fpair = (unsigned*)(ws + (2u << 20));            // 2 MB
    unsigned*       hpack = (unsigned*)(ws + (4u << 20));            // 2 MB
    float*          xmean = (float*)(ws + (6u << 20));               // 64 KB
    float*          mBuf  = (float*)(ws + (6u << 20) + (64u << 10)); // 64 KB
    float*          ima1  = (float*)(ws + (6u << 20) + (128u << 10));// 1 KB

    k_zero <<<1, 256, 0, stream>>>(ima1, 256);
    k_proj <<<256, 128, 0, stream>>>(x, wf, bfv, wg, bgv, wh, bhv,
                                     gBuf, fpair, hpack, xmean);
    k_attn <<<128, 256, 0, stream>>>(gBuf, fpair, hpack, wr, br, xmean, mBuf);
    k_ima1 <<<64, 256, 0, stream>>>(x, w1, ima1);
    k_final<<<1, 512, 0, stream>>>(ima1, b1, w2, b2, mBuf, wc1, bc1, wc2, bc2, out);
}
